// PatchSeparableBlockFactorizedConvolutionBlock_50852412785140
// MI455X (gfx1250) — matrix operand order repaired and checked by reading, not compiled, not run on hardware
//
#include <hip/hip_runtime.h>

typedef __attribute__((ext_vector_type(16))) _Float16 v16h;
typedef __attribute__((ext_vector_type(8)))  _Float16 v8h;
typedef __attribute__((ext_vector_type(8)))  float    v8f;

#define DX   2048
#define HDIM 64
#define W3N  1024   // C*HI*HO
#define DOUT 128
#define PNUM 2048

__device__ __forceinline__ v8f wmma_f16(v16h a, v16h b, v8f c) {
    // v_wmma_f32_16x16x32_f16: D = A(16x32 f16) x B(32x16 f16) + C(16x16 f32)
    return __builtin_amdgcn_wmma_f32_16x16x32_f16(false, a, false, b, (short)0, c, false, false);
}

// ---- CDNA5 async global->LDS copy (ASYNCcnt tracked) ----
__device__ __forceinline__ unsigned lds_off_of(const void* p) {
    // flat shared address: addr[31:0] is the LDS offset (ISA 10.2 aperture mapping)
    return (unsigned)(unsigned long long)p;
}
__device__ __forceinline__ void async_g2l_b128(unsigned lds_off, const void* g) {
    // per-lane 16B: LDS[lds_off(lane)] = MEM[g(lane)]
    asm volatile("global_load_async_to_lds_b128 %0, %1, off"
                 :: "v"(lds_off), "v"(g) : "memory");
}
template <int Nn>
__device__ __forceinline__ void wait_asynccnt() {
    asm volatile("s_wait_asynccnt %0" :: "n"(Nn) : "memory");
}
// copy BYTES contiguous bytes with 4 waves; BYTES/(4*512) async instrs per wave
template <int BYTES>
__device__ __forceinline__ void stage_region(const char* __restrict__ g, unsigned lds_off,
                                             int wave, int lane) {
    constexpr int PW = BYTES / (4 * 512);
#pragma unroll
    for (int t = 0; t < PW; ++t) {
        unsigned off = ((unsigned)(wave * PW + t) * 32u + (unsigned)lane) * 16u;
        async_g2l_b128(lds_off + off, g + off);
    }
}

// ---------------- weight swizzle: f32 row-major [K x Nn] -> f16 WMMA-B fragments ----------------
// B layout (16-bit, 32x16): lane L holds col = L&15; slot s holds K = 8*(L>>4) + (s&7) + 16*(s>>3).
__global__ void k_swz(const float* __restrict__ W, _Float16* __restrict__ out, int K, int Nn) {
    int idx = blockIdx.x * blockDim.x + threadIdx.x;
    int total = K * Nn;
    if (idx >= total) return;
    int NT   = Nn >> 4;
    int frag = idx >> 9;           // 512 elements per fragment
    int lane = (idx >> 4) & 31;
    int slot = idx & 15;
    int kb = frag / NT, jt = frag % NT;
    int Kk  = kb * 32 + 8 * (lane >> 4) + (slot & 7) + 16 * (slot >> 3);
    int col = jt * 16 + (lane & 15);
    out[idx] = (_Float16)W[(size_t)Kk * Nn + col];
}

__global__ void k_zero(float* __restrict__ p, int n) {
    int i = blockIdx.x * blockDim.x + threadIdx.x;
    if (i < n) p[i] = 0.0f;
}

__global__ void k_count(const int* __restrict__ batch, float* __restrict__ cnts, int n) {
    int i = blockIdx.x * blockDim.x + threadIdx.x;
    if (i < n) atomicAdd(&cnts[batch[i]], 1.0f);
}

__global__ void k_mean(const float* __restrict__ sums, const float* __restrict__ cnts,
                       float* __restrict__ vcy, int n) {
    int i = blockIdx.x * blockDim.x + threadIdx.x;
    if (i < n) vcy[i] = sums[i] / fmaxf(cnts[i >> 7], 1.0f);
}

// A fragment (16-bit, 16x32) from an f16 row-major [16 x 64] LDS tile, K base = kbase.
// Lane L: row = L&15; slot s: K = kbase + 8*(L>>4) + (s&7) + 16*(s>>3).
__device__ __forceinline__ v16h load_a_lds(const _Float16* __restrict__ h, int lane, int kbase) {
    int row = lane & 15, half = lane >> 4;
    int koff = kbase + 8 * half;
    v8h c0 = *(const v8h*)(h + row * 64 + koff);
    v8h c1 = *(const v8h*)(h + row * 64 + koff + 16);
    v16h a;
#pragma unroll
    for (int s = 0; s < 8; ++s) { a[s] = c0[s]; a[s + 8] = c1[s]; }
    return a;
}

// ---------------- fused dual-MLP: wy -> (v . wy) segment-sum atomics ; wx -> f16 scratch ----------
__global__ __launch_bounds__(128) void k_mlp(
    const float* __restrict__ x, const float* __restrict__ v, const int* __restrict__ batch,
    const _Float16* __restrict__ w1a, const float* __restrict__ b1a,
    const _Float16* __restrict__ w2a, const float* __restrict__ b2a,
    const _Float16* __restrict__ w3a, const float* __restrict__ b3a,
    const _Float16* __restrict__ w1b, const float* __restrict__ b1b,
    const _Float16* __restrict__ w2b, const float* __restrict__ b2b,
    const _Float16* __restrict__ w3b, const float* __restrict__ b3b,
    float* __restrict__ sums, _Float16* __restrict__ wx, int n)
{
    __shared__ __align__(16) _Float16 sh_stage[2][8192];   // 2 x 16KB double-buffered B stage
    __shared__ __align__(16) _Float16 sh_h1a[4][16 * 64];
    __shared__ __align__(16) _Float16 sh_h1b[4][16 * 64];
    __shared__ __align__(16) _Float16 sh_h2 [4][16 * 64];
    __shared__ __align__(16) float    sh_kyv[4][16 * 128];
    __shared__ __align__(16) float    sh_vt [4][16 * 128];

    const int lane = threadIdx.x & 31;
    const int wave = threadIdx.x >> 5;
    const int m0   = (blockIdx.x * 4 + wave) * 16;
    const bool valid = (m0 < n);           // wave-uniform; N % 16 == 0
    const int  mR = valid ? m0 : (n - 16); // clamped base for loads only
    const int half = lane >> 4, r15 = lane & 15;

    const unsigned st0 = lds_off_of(&sh_stage[0][0]);
    const unsigned st1 = lds_off_of(&sh_stage[1][0]);

    // ---- layer 1 for BOTH weight sets (x read once; B staged via async-to-LDS) ----
    stage_region<8192>((const char*)w1a, st0, wave, lane);
    stage_region<8192>((const char*)w1b, st0 + 8192, wave, lane);

    v8f acc1a[4] = {}, acc1b[4] = {};
    const float* xrow = x + (size_t)(mR + r15) * DX + 8 * half;
    for (int c = 0; c < 32; ++c) {                 // chunk = 2 K-steps (16KB of B frags)
        if (c + 1 < 32) {
            unsigned boff = ((c + 1) & 1) ? st1 : st0;
            stage_region<8192>((const char*)w1a + (size_t)(c + 1) * 8192, boff, wave, lane);
            stage_region<8192>((const char*)w1b + (size_t)(c + 1) * 8192, boff + 8192, wave, lane);
            wait_asynccnt<8>();                    // retire chunk c, chunk c+1 in flight
        } else {
            wait_asynccnt<0>();
        }
        __syncthreads();
        const _Float16* bufa = sh_stage[c & 1];
        const _Float16* bufb = bufa + 4096;
#pragma unroll
        for (int kk = 0; kk < 2; ++kk) {
            const float* p0 = xrow + (c * 2 + kk) * 32;   // slots 0..7 : K..K+7
            const float* p1 = p0 + 16;                    // slots 8..15: K+16..K+23
            v16h a;
#pragma unroll
            for (int s = 0; s < 8; ++s) { a[s] = (_Float16)p0[s]; a[s + 8] = (_Float16)p1[s]; }
#pragma unroll
            for (int j = 0; j < 4; ++j) {
                v16h ba = *(const v16h*)(bufa + ((kk * 4 + j) * 32 + lane) * 16);
                acc1a[j] = wmma_f16(a, ba, acc1a[j]);
                v16h bb = *(const v16h*)(bufb + ((kk * 4 + j) * 32 + lane) * 16);
                acc1b[j] = wmma_f16(a, bb, acc1b[j]);
            }
        }
        __syncthreads();
    }

    // bias + relu -> LDS (f16); C/D layout: VGPR r, lane L -> (m = r+8*(L>>4), col = j*16+(L&15))
#pragma unroll
    for (int j = 0; j < 4; ++j) {
        int col = j * 16 + r15;
        float bA = b1a[col], bB = b1b[col];
#pragma unroll
        for (int r = 0; r < 8; ++r) {
            int m = r + 8 * half;
            float ha = acc1a[j][r] + bA; ha = ha > 0.0f ? ha : 0.0f;
            float hb = acc1b[j][r] + bB; hb = hb > 0.0f ? hb : 0.0f;
            sh_h1a[wave][m * 64 + col] = (_Float16)ha;
            sh_h1b[wave][m * 64 + col] = (_Float16)hb;
        }
    }

    // stage v tile, zero ky_v accumulator (per-wave regions)
    for (int i = lane; i < 16 * 128; i += 32) {
        sh_vt [wave][i] = v[(size_t)mR * 128 + i];
        sh_kyv[wave][i] = 0.0f;
    }

    // ---- per set: layer 2 (direct), layer 3 (staged), consume ----
    for (int set = 0; set < 2; ++set) {
        const _Float16* h1  = set == 0 ? sh_h1a[wave] : sh_h1b[wave];
        const _Float16* w2s = set == 0 ? w2a : w2b;
        const float*    b2  = set == 0 ? b2a : b2b;
        const _Float16* w3s = set == 0 ? w3a : w3b;
        const float*    b3  = set == 0 ? b3a : b3b;

        // layer 2: [16,64] = relu([16,64] @ [64,64] + b2)   (w2 is tiny: direct loads)
        v8f acc2[4] = {};
#pragma unroll
        for (int ka = 0; ka < 2; ++ka) {
            v16h a = load_a_lds(h1, lane, ka * 32);
#pragma unroll
            for (int j = 0; j < 4; ++j) {
                v16h b = *(const v16h*)(w2s + ((size_t)(ka * 4 + j) * 32 + lane) * 16);
                acc2[j] = wmma_f16(a, b, acc2[j]);
            }
        }
        _Float16* h2 = sh_h2[wave];
#pragma unroll
        for (int j = 0; j < 4; ++j) {
            int col = j * 16 + r15;
            float bB = b2[col];
#pragma unroll
            for (int r = 0; r < 8; ++r) {
                int m = r + 8 * half;
                float hv = acc2[j][r] + bB; hv = hv > 0.0f ? hv : 0.0f;
                h2[m * 64 + col] = (_Float16)hv;
            }
        }

        // layer 3: [16,1024] as 64 N-tiles; stage 8 tiles x 2 K-frags per chunk (16KB)
        v16h a0 = load_a_lds(h2, lane, 0);
        v16h a1 = load_a_lds(h2, lane, 32);
        // w3 swizzle layout: kb0 frags at bytes [0,64K), kb1 frags at [64K,128K)
        stage_region<8192>((const char*)w3s,         st0,        wave, lane);
        stage_region<8192>((const char*)w3s + 65536, st0 + 8192, wave, lane);
        for (int ch = 0; ch < 8; ++ch) {
            if (ch + 1 < 8) {
                unsigned boff = ((ch + 1) & 1) ? st1 : st0;
                stage_region<8192>((const char*)w3s + (size_t)(ch + 1) * 8192, boff, wave, lane);
                stage_region<8192>((const char*)w3s + 65536 + (size_t)(ch + 1) * 8192, boff + 8192, wave, lane);
                wait_asynccnt<8>();
            } else {
                wait_asynccnt<0>();
            }
            __syncthreads();
            const _Float16* buf = sh_stage[ch & 1];
#pragma unroll
            for (int jj = 0; jj < 8; ++jj) {
                int j = ch * 8 + jj;
                v8f acc3 = {};
                v16h bt0 = *(const v16h*)(buf + (jj * 32 + lane) * 16);
                acc3 = wmma_f16(a0, bt0, acc3);
                v16h bt1 = *(const v16h*)(buf + 4096 + (jj * 32 + lane) * 16);
                acc3 = wmma_f16(a1, bt1, acc3);
                int col = j * 16 + r15;
                float bB = b3[col];
                if (set == 0) {
                    // wy tile -> ky_v[m, c*8+ho] += v[m, c*8+hi] * wy ; col = c*64 + hi*8 + ho
                    int cc = j >> 2;
                    int hi = 2 * (j & 3) + (r15 >> 3);
                    int ho = r15 & 7;
                    if (valid) {
#pragma unroll
                        for (int r = 0; r < 8; ++r) {
                            int m = r + 8 * half;
                            float val = acc3[r] + bB;
                            atomicAdd(&sh_kyv[wave][m * 128 + cc * 8 + ho],
                                      sh_vt[wave][m * 128 + cc * 8 + hi] * val);
                        }
                    }
                } else {
                    if (valid) {
#pragma unroll
                        for (int r = 0; r < 8; ++r) {
                            int m = r + 8 * half;
                            wx[(size_t)(m0 + m) * W3N + col] = (_Float16)(acc3[r] + bB);
                        }
                    }
                }
            }
            __syncthreads();
        }
    }

    // flush ky_v tile -> global segment sums (f32 atomics)
    if (valid) {
        for (int i = lane; i < 16 * 128; i += 32) {
            int m = i >> 7, d = i & 127;
            int p = batch[m0 + m];
            atomicAdd(&sums[(size_t)p * 128 + d], sh_kyv[wave][i]);
        }
    }
}

// ---------------- gather + second blockwise einsum + skip-linear (WMMA) ----------------
__global__ __launch_bounds__(128) void k_out(
    const float* __restrict__ v, const int* __restrict__ batch,
    const float* __restrict__ vcy, const _Float16* __restrict__ wx,
    const _Float16* __restrict__ linsw, const float* __restrict__ lin_b,
    float* __restrict__ out, int n)
{
    __shared__ __align__(16) float sh_g [4][16 * 128];
    __shared__ __align__(16) float sh_vc[4][16 * 128];

    const int lane = threadIdx.x & 31;
    const int wave = threadIdx.x >> 5;
    const int m0   = (blockIdx.x * 4 + wave) * 16;
    if (m0 >= n) return;                   // no barriers in this kernel: early-exit is safe
    const int half = lane >> 4, r15 = lane & 15;

    // gather g[m] = v_conv_y[batch[m]]
    for (int i = lane; i < 16 * 128; i += 32) {
        int m = i >> 7, d = i & 127;
        sh_g[wave][i] = vcy[(size_t)batch[m0 + m] * 128 + d];
    }

    // v_conv[m, c*8+p] = sum_ho g[m, c*8+ho] * wx[m, c*64+ho*8+p]
    for (int t = 0; t < 8; ++t) {
        int pair = lane + 32 * t;          // (m,c) pair, 256 total
        int m = pair >> 4, c = pair & 15;
        const _Float16* wrow = wx + (size_t)(m0 + m) * W3N + c * 64;
        float acc[8] = {0, 0, 0, 0, 0, 0, 0, 0};
#pragma unroll
        for (int ho = 0; ho < 8; ++ho) {
            float gv = sh_g[wave][m * 128 + c * 8 + ho];
            v8h wv = *(const v8h*)(wrow + ho * 8);
#pragma unroll
            for (int p = 0; p < 8; ++p) acc[p] += gv * (float)wv[p];
        }
#pragma unroll
        for (int p = 0; p < 8; ++p) sh_vc[wave][m * 128 + c * 8 + p] = acc[p];
    }

    // out = v + v_conv @ lin_w + lin_b  (WMMA, K=128, 8 N-tiles)
    v8f acc[8] = {};
#pragma unroll
    for (int ka = 0; ka < 4; ++ka) {
        int koff = ka * 32 + 8 * half;
        const float* row = sh_vc[wave] + r15 * 128;
        v16h a;
#pragma unroll
        for (int s = 0; s < 8; ++s) {
            a[s]     = (_Float16)row[koff + s];
            a[s + 8] = (_Float16)row[koff + 16 + s];
        }
#pragma unroll
        for (int j = 0; j < 8; ++j) {
            v16h b = *(const v16h*)(linsw + ((size_t)(ka * 8 + j) * 32 + lane) * 16);
            acc[j] = wmma_f16(a, b, acc[j]);
        }
    }
#pragma unroll
    for (int j = 0; j < 8; ++j) {
        int col = j * 16 + r15;
        float bB = lin_b[col];
#pragma unroll
        for (int r = 0; r < 8; ++r) {
            int m = r + 8 * half;
            size_t o = (size_t)(m0 + m) * 128 + col;
            out[o] = v[o] + acc[j][r] + bB;
        }
    }
}

extern "C" void kernel_launch(void* const* d_in, const int* in_sizes, int n_in,
                              void* d_out, int out_size, void* d_ws, size_t ws_size,
                              hipStream_t stream) {
    const float* x     = (const float*)d_in[0];
    const float* v     = (const float*)d_in[1];
    const int*   batch = (const int*)  d_in[2];
    // d_in[3] = num_patches (device scalar) -> fixed at 2048 per reference
    const float* k1_w1 = (const float*)d_in[4];  const float* k1_b1 = (const float*)d_in[5];
    const float* k1_w2 = (const float*)d_in[6];  const float* k1_b2 = (const float*)d_in[7];
    const float* k1_w3 = (const float*)d_in[8];  const float* k1_b3 = (const float*)d_in[9];
    const float* k2_w1 = (const float*)d_in[10]; const float* k2_b1 = (const float*)d_in[11];
    const float* k2_w2 = (const float*)d_in[12]; const float* k2_b2 = (const float*)d_in[13];
    const float* k2_w3 = (const float*)d_in[14]; const float* k2_b3 = (const float*)d_in[15];
    const float* lin_w = (const float*)d_in[16]; const float* lin_b = (const float*)d_in[17];
    float* out = (float*)d_out;
    const int n = in_sizes[2];                   // N (multiple of 16 per reference)

    // workspace layout (256B aligned chunks)
    char* ws = (char*)d_ws;
    size_t off = 0;
    auto take = [&](size_t bytes) { char* p = ws + off; off = (off + bytes + 255) & ~(size_t)255; return p; };
    _Float16* w1a  = (_Float16*)take((size_t)DX * HDIM * 2);
    _Float16* w1b  = (_Float16*)take((size_t)DX * HDIM * 2);
    _Float16* w2a  = (_Float16*)take((size_t)HDIM * HDIM * 2);
    _Float16* w2b  = (_Float16*)take((size_t)HDIM * HDIM * 2);
    _Float16* w3a  = (_Float16*)take((size_t)HDIM * W3N * 2);
    _Float16* w3b  = (_Float16*)take((size_t)HDIM * W3N * 2);
    _Float16* lins = (_Float16*)take((size_t)DOUT * DOUT * 2);
    float*    sums = (float*)take((size_t)PNUM * 128 * 4 + (size_t)PNUM * 4);
    float*    cnts = sums + (size_t)PNUM * 128;          // contiguous: one zero pass
    float*    vcy  = (float*)take((size_t)PNUM * 128 * 4);
    _Float16* wx   = (_Float16*)take((size_t)n * W3N * 2);

    // 1) swizzle all weights to f16 WMMA-B fragments
    k_swz<<<(DX * HDIM + 255) / 256, 256, 0, stream>>>(k1_w1, w1a, DX, HDIM);
    k_swz<<<(DX * HDIM + 255) / 256, 256, 0, stream>>>(k2_w1, w1b, DX, HDIM);
    k_swz<<<(HDIM * HDIM + 255) / 256, 256, 0, stream>>>(k1_w2, w2a, HDIM, HDIM);
    k_swz<<<(HDIM * HDIM + 255) / 256, 256, 0, stream>>>(k2_w2, w2b, HDIM, HDIM);
    k_swz<<<(HDIM * W3N + 255) / 256, 256, 0, stream>>>(k1_w3, w3a, HDIM, W3N);
    k_swz<<<(HDIM * W3N + 255) / 256, 256, 0, stream>>>(k2_w3, w3b, HDIM, W3N);
    k_swz<<<(DOUT * DOUT + 255) / 256, 256, 0, stream>>>(lin_w, lins, DOUT, DOUT);

    // 2) zero segment sums + counts; count nodes per patch
    const int nz = PNUM * 128 + PNUM;
    k_zero<<<(nz + 255) / 256, 256, 0, stream>>>(sums, nz);
    k_count<<<(n + 255) / 256, 256, 0, stream>>>(batch, cnts, n);

    // 3) fused dual-MLP + first einsum + segment-sum atomics; wx to scratch
    const int blocks = (n + 63) / 64;
    k_mlp<<<blocks, 128, 0, stream>>>(x, v, batch,
                                      w1a, k1_b1, w2a, k1_b2, w3a, k1_b3,
                                      w1b, k2_b1, w2b, k2_b2, w3b, k2_b3,
                                      sums, wx, n);

    // 4) segment mean
    k_mean<<<(PNUM * 128 + 255) / 256, 256, 0, stream>>>(sums, cnts, vcy, PNUM * 128);

    // 5) gather + second einsum + skip-linear
    k_out<<<blocks, 128, 0, stream>>>(v, batch, vcy, wx, lins, lin_b, out, n);
}
